// ReceptorConv_64982855188920
// MI455X (gfx1250) — compile-verified, matrix operand order checked by reading, not run on hardware
//
#include <hip/hip_runtime.h>
#include <hip/hip_bf16.h>
#include <stdint.h>

// ---------------------------------------------------------------------------
// EGNN layer for MI455X (gfx1250, wave32, WMMA).
// v5: spill-free latency hiding. Same blocking as v4 (64 edges/block, split
// eW1/cW1 k-passes, one 64-VGPR accumulator set, 8 KB LDS B staging shared by
// 4 waves, ~2.5 KB/edge B traffic). Changes:
//  - Dropped the global->reg staging prefetch (it was spilled to scratch);
//    staging is a plain global->LDS copy in the barrier window, hidden by
//    other resident blocks.
//  - The 8-WMMA chunk is hand-unrolled with two alternating B buffers whose
//    live ranges overlap at every point, so the allocator cannot coalesce
//    them and each B LDS load pair has a full WMMA between issue and use
//    (partial s_wait_dscnt instead of full drains).
// ---------------------------------------------------------------------------

typedef __attribute__((ext_vector_type(16))) _Float16 v16h;
typedef __attribute__((ext_vector_type(8)))  float    v8f;
typedef __attribute__((ext_vector_type(4)))  _Float16 v4h;

#define WPB 4          // waves per block (128 threads)
#define FEAT 128

union FragU { v16h v; uint4 u[2]; };

__device__ __forceinline__ v16h lds_frag(const _Float16* rowBase, int byteOff) {
  // A fragment (16-bit, 16x32): halves 0..7  = row bytes [off, off+16)
  //                             halves 8..15 = row bytes [off+32, off+48)
  FragU f;
  const char* p = (const char*)rowBase + byteOff;
  f.u[0] = *(const uint4*)(p);
  f.u[1] = *(const uint4*)(p + 32);
  return f.v;
}

__device__ __forceinline__ v16h bfrag_at(const _Float16* base, int nt, int lane) {
  // packed B fragment: 512 halves per fragment, 32 B per lane
  FragU f;
  const uint4* q = (const uint4*)(base + nt * 512 + (lane << 4));
  f.u[0] = q[0]; f.u[1] = q[1];
  return f.v;
}

__device__ __forceinline__ float sigm(float x) { return 1.0f / (1.0f + __expf(-x)); }
__device__ __forceinline__ float silu(float x) { return x * sigm(x); }

__device__ __forceinline__ v8f wmma_f16(v16h a, v16h b, v8f c) {
  return __builtin_amdgcn_wmma_f32_16x16x32_f16(false, a, false, b, (short)0, c,
                                                false, false);
}

// --------------------------- utility kernels -------------------------------

__global__ void zero_kernel(float* __restrict__ p, size_t n) {
  size_t i = (size_t)blockIdx.x * blockDim.x + threadIdx.x;
  size_t stride = (size_t)gridDim.x * blockDim.x;
  for (; i < n; i += stride) p[i] = 0.0f;
}

// Pack W[Ktot x Nout] (row-major, fp32) into WMMA B fragments (fp16).
// Fragment (kc,nt) lives at out[(kc*NT+nt)*512 ...]; lane L, half h ->
// K = 32*kc + 16*(h/8) + 8*(L>=16) + (h%8), N = 16*nt + (L%16).
__global__ void pack_b_kernel(const float* __restrict__ W, _Float16* __restrict__ out,
                              int KC, int NT, int Nout) {
  int tid = blockIdx.x * blockDim.x + threadIdx.x;
  int total = KC * NT * 512;
  if (tid >= total) return;
  int frag = tid >> 9;
  int rem  = tid & 511;
  int lane = rem >> 4;
  int h    = rem & 15;
  int kc = frag / NT, nt = frag % NT;
  int K = kc * 32 + (h >> 3) * 16 + ((lane >= 16) ? 8 : 0) + (h & 7);
  int N = nt * 16 + (lane & 15);
  out[tid] = (_Float16)W[(size_t)K * Nout + N];
}

// ------------------------------ edge kernel --------------------------------

__global__ __launch_bounds__(128) void egnn_edge_kernel(
    const float* __restrict__ node_feat, const float* __restrict__ coord,
    const int* __restrict__ srcI, const int* __restrict__ dstI,
    const float* __restrict__ eW1, const float* __restrict__ eb1,
    const float* __restrict__ eb2,
    const float* __restrict__ aW,  const float* __restrict__ ab,
    const float* __restrict__ cW1, const float* __restrict__ cb1,
    const float* __restrict__ cW2,
    const _Float16* __restrict__ eW1p, const _Float16* __restrict__ eW2p,
    const _Float16* __restrict__ cW1p,
    float* __restrict__ h_acc, float* __restrict__ x_acc,
    int E, int nTiles)
{
  __shared__ alignas(16) _Float16 sA[WPB][16][264];   // A rows, 33792 B
  __shared__ alignas(16) _Float16 sM[WPB][16][136];   // msg1 rows, 17408 B
  __shared__ alignas(16) _Float16 sStage[4096];       // B staging, 8192 B
  __shared__ float sRad[WPB][16];
  __shared__ float sXd[WPB][16][3];
  __shared__ float sCo[WPB][16];
  __shared__ int   sDst[WPB][16];

  const int w    = threadIdx.x >> 5;
  const int lane = threadIdx.x & 31;
  const int ln   = lane & 15;
  const int hi   = lane >> 4;                 // 0: lanes 0-15, 1: lanes 16-31
  const int tile = blockIdx.x * WPB + w;
  const bool valid = (tile < nTiles);
  const int base = tile * 16;
  const int tix = threadIdx.x;

  v8f acc[8];   // one accumulator set, reused across the three GEMM passes

  // Staged GEMM pass: per 32-wide k-chunk, stage all 8 n-tiles of B into
  // sStage, then 8 WMMAs with two alternating (un-coalescable) B buffers.
  auto gemm_pass = [&](const _Float16* Bp, int KC, const _Float16* aBase) {
    for (int kc = 0; kc < KC; ++kc) {
      __syncthreads();                  // previous chunk's B reads complete
      {
        const uint4* g = (const uint4*)(Bp + kc * 4096);
        uint4* d = (uint4*)sStage;
        #pragma unroll
        for (int i = 0; i < 4; ++i) d[tix + i * 128] = g[tix + i * 128];
      }
      __syncthreads();
      if (valid) {
        v16h a  = lds_frag(aBase, kc * 64 + hi * 16);
        v16h b0 = bfrag_at(sStage, 0, lane);
        v16h b1 = bfrag_at(sStage, 1, lane);
        acc[0] = wmma_f16(a, b0, acc[0]); b0 = bfrag_at(sStage, 2, lane);
        acc[1] = wmma_f16(a, b1, acc[1]); b1 = bfrag_at(sStage, 3, lane);
        acc[2] = wmma_f16(a, b0, acc[2]); b0 = bfrag_at(sStage, 4, lane);
        acc[3] = wmma_f16(a, b1, acc[3]); b1 = bfrag_at(sStage, 5, lane);
        acc[4] = wmma_f16(a, b0, acc[4]); b0 = bfrag_at(sStage, 6, lane);
        acc[5] = wmma_f16(a, b1, acc[5]); b1 = bfrag_at(sStage, 7, lane);
        acc[6] = wmma_f16(a, b0, acc[6]);
        acc[7] = wmma_f16(a, b1, acc[7]);
      }
    }
  };

  // ---------------- stage: geometry + A rows ----------------
  if (valid) {
    if (lane < 16) {
      int e = base + lane; if (e >= E) e = E - 1;
      int s = srcI[e], d = dstI[e];
      sDst[w][lane] = d;
      float dx = coord[s * 3 + 0] - coord[d * 3 + 0];
      float dy = coord[s * 3 + 1] - coord[d * 3 + 1];
      float dz = coord[s * 3 + 2] - coord[d * 3 + 2];
      float r = sqrtf(dx * dx + dy * dy + dz * dz);
      sRad[w][lane] = r;
      float inv = 1.0f / (r + 1.0f);
      sXd[w][lane][0] = dx * inv;
      sXd[w][lane][1] = dy * inv;
      sXd[w][lane][2] = dz * inv;
    }
    {
      int row = lane >> 1, part = lane & 1;
      int e = base + row; if (e >= E) e = E - 1;
      int idx = part ? dstI[e] : srcI[e];
      const float4* srcp = (const float4*)(node_feat + (size_t)idx * FEAT);
      _Float16* drow = &sA[w][row][part * FEAT];
      #pragma unroll
      for (int i = 0; i < 32; ++i) {
        float4 f = srcp[i];
        v4h hv; hv.x = (_Float16)f.x; hv.y = (_Float16)f.y;
                hv.z = (_Float16)f.z; hv.w = (_Float16)f.w;
        *(v4h*)(drow + i * 4) = hv;
      }
    }
  }
  __syncthreads();

  const _Float16* aRow = &sA[w][ln][0];
  const _Float16* mRow = &sM[w][ln][0];
  float radm[8];
  if (valid) {
    #pragma unroll
    for (int v = 0; v < 8; ++v) radm[v] = sRad[w][v + hi * 8];
  }

  // ---------------- pass 1a: f @ eW1 (+ bias + radial row) ----------------
  if (valid) {
    #pragma unroll
    for (int nt = 0; nt < 8; ++nt) {
      const int n = nt * 16 + ln;
      const float be = eb1[n];
      const float we = eW1[256 * FEAT + n];   // radial row folded into C init
      #pragma unroll
      for (int v = 0; v < 8; ++v) acc[nt][v] = be + radm[v] * we;
    }
  }
  gemm_pass(eW1p, 8, aRow);
  if (valid) {                     // msg1 = silu(.) -> sM (per-wave slice)
    #pragma unroll
    for (int nt = 0; nt < 8; ++nt) {
      const int n = nt * 16 + ln;
      #pragma unroll
      for (int v = 0; v < 8; ++v) sM[w][v + hi * 8][n] = (_Float16)silu(acc[nt][v]);
    }
  }

  // ---------------- pass 1b: f @ cW1 -> coord_out ----------------
  if (valid) {
    #pragma unroll
    for (int nt = 0; nt < 8; ++nt) {
      const int n = nt * 16 + ln;
      const float bc = cb1[n];
      const float wc = cW1[256 * FEAT + n];
      #pragma unroll
      for (int v = 0; v < 8; ++v) acc[nt][v] = bc + radm[v] * wc;
    }
  }
  gemm_pass(cW1p, 8, aRow);
  if (valid) {
    float pC[8];
    #pragma unroll
    for (int v = 0; v < 8; ++v) pC[v] = 0.0f;
    #pragma unroll
    for (int nt = 0; nt < 8; ++nt) {
      const float cw2 = cW2[nt * 16 + ln];
      #pragma unroll
      for (int v = 0; v < 8; ++v) pC[v] += silu(acc[nt][v]) * cw2;
    }
    #pragma unroll
    for (int v = 0; v < 8; ++v) {
      #pragma unroll
      for (int m = 8; m >= 1; m >>= 1) pC[v] += __shfl_xor(pC[v], m, 16);
    }
    if (ln == 0) {
      #pragma unroll
      for (int v = 0; v < 8; ++v) sCo[w][v + hi * 8] = pC[v];
    }
  }
  __syncthreads();                      // sCo (and msg1) visible

  // ---------------- coord scatter ----------------
  if (valid && lane < 16 && (base + lane) < E) {
    float co = tanhf(sCo[w][lane]) * 10.0f;   // COORDS_RANGE
    int d = sDst[w][lane];
    atomicAdd(&x_acc[d * 3 + 0], co * sXd[w][lane][0]);
    atomicAdd(&x_acc[d * 3 + 1], co * sXd[w][lane][1]);
    atomicAdd(&x_acc[d * 3 + 2], co * sXd[w][lane][2]);
  }

  // ---------------- pass 2: msg1 @ eW2 ----------------
  if (valid) {
    #pragma unroll
    for (int nt = 0; nt < 8; ++nt) {
      const float b2 = eb2[nt * 16 + ln];
      #pragma unroll
      for (int v = 0; v < 8; ++v) acc[nt][v] = b2;
    }
  }
  gemm_pass(eW2p, 4, mRow);

  // ---------------- SiLU, aW gate, message scatter ----------------
  if (valid) {
    int dstm[8]; bool rowOk[8];
    #pragma unroll
    for (int v = 0; v < 8; ++v) {
      int m = v + hi * 8;
      dstm[v] = sDst[w][m];
      rowOk[v] = (base + m) < E;
    }
    float pg[8];
    #pragma unroll
    for (int v = 0; v < 8; ++v) pg[v] = 0.0f;
    #pragma unroll
    for (int nt = 0; nt < 8; ++nt) {
      const float awn = aW[nt * 16 + ln];
      #pragma unroll
      for (int v = 0; v < 8; ++v) {
        float h2 = silu(acc[nt][v]);
        acc[nt][v] = h2;                 // keep ungated value in regs
        pg[v] += h2 * awn;
      }
    }
    const float abv = ab[0];
    #pragma unroll
    for (int v = 0; v < 8; ++v) {
      #pragma unroll
      for (int m = 8; m >= 1; m >>= 1) pg[v] += __shfl_xor(pg[v], m, 16);
      pg[v] = sigm(pg[v] + abv);          // per-row gate
    }
    #pragma unroll
    for (int nt = 0; nt < 8; ++nt) {
      const int n = nt * 16 + ln;
      #pragma unroll
      for (int v = 0; v < 8; ++v) {
        if (rowOk[v])
          atomicAdd(&h_acc[(size_t)dstm[v] * FEAT + n], acc[nt][v] * pg[v]);
      }
    }
  }
}

// ------------------------------ node kernel --------------------------------

__global__ __launch_bounds__(128) void egnn_node_kernel(
    const float* __restrict__ node_feat, const float* __restrict__ coord,
    const float* __restrict__ z,
    const float* __restrict__ nb1, const float* __restrict__ nb2,
    const _Float16* __restrict__ nW1p, const _Float16* __restrict__ nW2p,
    const float* __restrict__ h_acc, const float* __restrict__ x_acc,
    float* __restrict__ out_h, float* __restrict__ out_x,
    int Nn, int nTiles)
{
  __shared__ alignas(16) _Float16 sA[WPB][16][264];  // [node_feat | h_neigh/z]
  __shared__ alignas(16) _Float16 sM[WPB][16][136];

  const int w    = threadIdx.x >> 5;
  const int lane = threadIdx.x & 31;
  const int ln   = lane & 15;
  const int hi   = lane >> 4;
  const int tile = blockIdx.x * WPB + w;
  const bool valid = (tile < nTiles);
  const int base = tile * 16;

  if (valid) {
    int row = lane >> 1, part = lane & 1;
    int nd = base + row; if (nd >= Nn) nd = Nn - 1;
    float scale = part ? (1.0f / z[nd]) : 1.0f;
    const float4* srcp =
        (const float4*)((part ? h_acc : node_feat) + (size_t)nd * FEAT);
    _Float16* drow = &sA[w][row][part * FEAT];
    #pragma unroll
    for (int i = 0; i < 32; ++i) {
      float4 f = srcp[i];
      v4h hv; hv.x = (_Float16)(f.x * scale); hv.y = (_Float16)(f.y * scale);
              hv.z = (_Float16)(f.z * scale); hv.w = (_Float16)(f.w * scale);
      *(v4h*)(drow + i * 4) = hv;
    }
  }
  __syncthreads();

  if (valid) {
    const _Float16* aRow = &sA[w][ln][0];
    for (int nt = 0; nt < 8; ++nt) {
      const int n = nt * 16 + ln;
      const float b1 = nb1[n];
      v8f acc;
      #pragma unroll
      for (int v = 0; v < 8; ++v) acc[v] = b1;
      #pragma unroll
      for (int kc = 0; kc < 8; ++kc) {
        v16h a = lds_frag(aRow, kc * 64 + hi * 16);
        FragU f;
        const uint4* q = (const uint4*)(nW1p + ((kc * 8 + nt) << 9) + (lane << 4));
        f.u[0] = q[0]; f.u[1] = q[1];
        acc = wmma_f16(a, f.v, acc);
      }
      #pragma unroll
      for (int v = 0; v < 8; ++v) sM[w][v + hi * 8][n] = (_Float16)silu(acc[v]);
    }
  }
  __syncthreads();

  if (valid) {
    const _Float16* mRow = &sM[w][ln][0];
    for (int nt = 0; nt < 8; ++nt) {
      const int n = nt * 16 + ln;
      const float b2 = nb2[n];
      v8f acc;
      #pragma unroll
      for (int v = 0; v < 8; ++v) acc[v] = b2;
      #pragma unroll
      for (int kc = 0; kc < 4; ++kc) {
        v16h a = lds_frag(mRow, kc * 64 + hi * 16);
        FragU f;
        const uint4* q = (const uint4*)(nW2p + ((kc * 8 + nt) << 9) + (lane << 4));
        f.u[0] = q[0]; f.u[1] = q[1];
        acc = wmma_f16(a, f.v, acc);
      }
      #pragma unroll
      for (int v = 0; v < 8; ++v) {
        int nd = base + v + hi * 8;
        if (nd < Nn) out_h[(size_t)nd * FEAT + n] = acc[v];
      }
    }
    if (lane < 16) {
      int nd = base + lane;
      if (nd < Nn) {
        float invz = 1.0f / z[nd];
        out_x[nd * 3 + 0] = coord[nd * 3 + 0] + x_acc[nd * 3 + 0] * invz;
        out_x[nd * 3 + 1] = coord[nd * 3 + 1] + x_acc[nd * 3 + 1] * invz;
        out_x[nd * 3 + 2] = coord[nd * 3 + 2] + x_acc[nd * 3 + 2] * invz;
      }
    }
  }
}

// ------------------------------- launcher ----------------------------------

extern "C" void kernel_launch(void* const* d_in, const int* in_sizes, int n_in,
                              void* d_out, int out_size, void* d_ws, size_t ws_size,
                              hipStream_t stream) {
  const float* node_feat = (const float*)d_in[0];
  const float* coord     = (const float*)d_in[1];
  const float* z         = (const float*)d_in[2];
  const int*   src       = (const int*)d_in[3];
  const int*   dst       = (const int*)d_in[4];
  const float* eW1 = (const float*)d_in[5];
  const float* eb1 = (const float*)d_in[6];
  const float* eW2 = (const float*)d_in[7];
  const float* eb2 = (const float*)d_in[8];
  const float* aW  = (const float*)d_in[9];
  const float* ab  = (const float*)d_in[10];
  const float* cW1 = (const float*)d_in[11];
  const float* cb1 = (const float*)d_in[12];
  const float* cW2 = (const float*)d_in[13];
  const float* nW1 = (const float*)d_in[14];
  const float* nb1 = (const float*)d_in[15];
  const float* nW2 = (const float*)d_in[16];
  const float* nb2 = (const float*)d_in[17];

  const int N = in_sizes[0] / FEAT;
  const int E = in_sizes[3];

  // ---- workspace layout ----
  uintptr_t p = (uintptr_t)d_ws;
  float* h_acc = (float*)p; p += (size_t)N * FEAT * sizeof(float);
  float* x_acc = (float*)p; p += (size_t)N * 3 * sizeof(float);
  p = (p + 255) & ~(uintptr_t)255;
  _Float16* eW1p = (_Float16*)p; p += 256 * 128 * sizeof(_Float16);
  _Float16* cW1p = (_Float16*)p; p += 256 * 128 * sizeof(_Float16);
  _Float16* eW2p = (_Float16*)p; p += 128 * 128 * sizeof(_Float16);
  _Float16* nW1p = (_Float16*)p; p += 256 * 128 * sizeof(_Float16);
  _Float16* nW2p = (_Float16*)p; p += 128 * 128 * sizeof(_Float16);

  // zero accumulators (h_acc and x_acc are contiguous: N*131 floats)
  zero_kernel<<<2048, 256, 0, stream>>>(h_acc, (size_t)N * (FEAT + 3));

  // pack weights into WMMA B fragments (fp16)
  pack_b_kernel<<<(8 * 8 * 512 + 255) / 256, 256, 0, stream>>>(eW1, eW1p, 8, 8, FEAT);
  pack_b_kernel<<<(8 * 8 * 512 + 255) / 256, 256, 0, stream>>>(cW1, cW1p, 8, 8, FEAT);
  pack_b_kernel<<<(4 * 8 * 512 + 255) / 256, 256, 0, stream>>>(eW2, eW2p, 4, 8, FEAT);
  pack_b_kernel<<<(8 * 8 * 512 + 255) / 256, 256, 0, stream>>>(nW1, nW1p, 8, 8, FEAT);
  pack_b_kernel<<<(4 * 8 * 512 + 255) / 256, 256, 0, stream>>>(nW2, nW2p, 4, 8, FEAT);

  // edge pipeline
  const int eTiles = (E + 15) / 16;
  egnn_edge_kernel<<<(eTiles + WPB - 1) / WPB, 128, 0, stream>>>(
      node_feat, coord, src, dst, eW1, eb1, eb2, aW, ab, cW1, cb1, cW2,
      eW1p, eW2p, cW1p, h_acc, x_acc, E, eTiles);

  // node update
  float* out_h = (float*)d_out;
  float* out_x = out_h + (size_t)N * FEAT;
  const int nTiles = (N + 15) / 16;
  egnn_node_kernel<<<(nTiles + WPB - 1) / WPB, 128, 0, stream>>>(
      node_feat, coord, z, nb1, nb2, nW1p, nW2p, h_acc, x_acc,
      out_h, out_x, N, nTiles);
}
